// GGNNMsgPass_1881195675812
// MI455X (gfx1250) — matrix operand order repaired and checked by reading, not compiled, not run on hardware
//
#include <hip/hip_runtime.h>

// ---------------------------------------------------------------------------
// GGNN message passing, fused, for MI455X (gfx1250, wave32, WMMA + TDM).
//
//   messages[b,i,m] = sum_j adj_w[bfm[b,i,j]-1, m, :] . afm[b,j,:] + bias[m]
//
// prep kernel : afm [B,N,NF] f32  ->  afmT16 [B,NF,N] fp16  (1 MB in d_ws)
// main kernel : per workgroup (b, 16-row tile), 8 waves = 8 edge types.
//   Phase 1: one-hot aggregation as dense f16 WMMA; A (one-hot) built in
//            registers from bfm, B tiles DMA'd into LDS by the Tensor Data
//            Mover (tensor_load_to_lds, double buffered, s_wait_tensorcnt).
//   Phase 2: 16x1024x16 GEMM vs adj_w; K = (e,n) sums over edge types.
//   agg never touches HBM: 32 KB of LDS between the phases.
// ---------------------------------------------------------------------------

typedef __attribute__((ext_vector_type(16))) _Float16 v16h;
typedef __attribute__((ext_vector_type(8)))  _Float16 v8h;
typedef __attribute__((ext_vector_type(8)))  float    v8f;
typedef __attribute__((ext_vector_type(4)))  unsigned int v4u;
typedef __attribute__((ext_vector_type(8)))  int      v8i;
typedef __attribute__((ext_vector_type(4)))  int      v4i;

#define NF 128   // node features
#define EF 8     // edge types (== waves per workgroup)
#define MF 128   // message features
#define BB 8     // batch
#define NN 512   // nodes
#define IT 16    // output rows per workgroup (WMMA M)
#define JT 32    // j-chunk (WMMA K for f16)
#define NCHUNK (NN / JT)

static_assert(EF == 8, "wave-per-edge-type mapping assumes EF==8");

#if defined(__AMDGCN__) && defined(__has_builtin)
#if __has_builtin(__builtin_amdgcn_tensor_load_to_lds) && \
    __has_builtin(__builtin_amdgcn_s_wait_tensorcnt)
#define USE_TDM 1
#endif
#endif
#ifndef USE_TDM
#define USE_TDM 0
#endif

__device__ __forceinline__ v16h cat16(v8h lo, v8h hi) {
  return __builtin_shufflevector(lo, hi, 0, 1, 2, 3, 4, 5, 6, 7,
                                         8, 9, 10, 11, 12, 13, 14, 15);
}

#if USE_TDM
// Issue a 2D TDM tile load: tile_d1 rows x tile_d0 elements from a row-major
// tensor (row stride = stride0 elements) into contiguous LDS at lds_off.
// D# layout per CDNA5 ISA 8.3/8.4: group0 = {count|flags, lds_addr, ga_lo,
// ga_hi|type=2}; group1 packs data_size, tensor_dim0/1, tile_dim0/1/2,
// tensor_dim0_stride. Groups 2/3 (and trailing group arg) zero: tensor is 2D.
// This toolchain exposes the 6-argument builtin form.
__device__ __forceinline__ void tdm_load_2d(unsigned lds_off, const void* tile_gaddr,
                                            unsigned tensor_d0, unsigned tensor_d1,
                                            unsigned tile_d0, unsigned tile_d1,
                                            unsigned stride0, unsigned ds_log2) {
  unsigned long long ga = (unsigned long long)tile_gaddr;
  v4u g0;
  g0[0] = 1u;                                   // count=1, user descriptor
  g0[1] = lds_off;                              // lds_addr (bytes)
  g0[2] = (unsigned)ga;                         // global_addr[31:0]
  g0[3] = (unsigned)(ga >> 32) | (2u << 30);    // global_addr[56:32] | type=2
  v8i g1;
  g1[0] = (int)(ds_log2 << 16);                 // data_size; mask/pad/flags=0
  g1[1] = (int)((tensor_d0 & 0xFFFFu) << 16);   // abar_addr=0 | tensor_dim0.lo
  g1[2] = (int)((tensor_d0 >> 16) | ((tensor_d1 & 0xFFFFu) << 16));
  g1[3] = (int)((tensor_d1 >> 16) | (tile_d0 << 16));
  g1[4] = (int)tile_d1;                         // tile_dim1 | tile_dim2=0
  g1[5] = (int)stride0;                         // tensor_dim0_stride[31:0]
  g1[6] = 0;                                    // stride0.hi | stride1.lo
  g1[7] = 0;
  v4i z4 = {0, 0, 0, 0};
  v8i z8 = {0, 0, 0, 0, 0, 0, 0, 0};
  __builtin_amdgcn_tensor_load_to_lds(g0, g1, z4, z4, z8, 0);
}
#endif

// ---------------------------------------------------------------------------
// prep: transpose + convert afm -> afmT16[b][n][j] fp16 (one-time, 3 MB I/O)
// ---------------------------------------------------------------------------
__global__ __launch_bounds__(256)
void ggnn_prep(const float* __restrict__ afm, _Float16* __restrict__ afmT16) {
  const int idx4 = blockIdx.x * 256 + threadIdx.x;  // float4 index in [B,N,NF]
  const int n4 = idx4 & (NF / 4 - 1);
  const int j  = (idx4 / (NF / 4)) & (NN - 1);
  const int b  = idx4 / ((NF / 4) * NN);
  const float4 v = *reinterpret_cast<const float4*>(afm + (size_t)idx4 * 4);
  _Float16* dst = afmT16 + ((size_t)b * NF + n4 * 4) * NN + j;
  dst[0 * NN] = (_Float16)v.x;
  dst[1 * NN] = (_Float16)v.y;
  dst[2 * NN] = (_Float16)v.z;
  dst[3 * NN] = (_Float16)v.w;
}

// ---------------------------------------------------------------------------
// main kernel: TDM-staged, double-buffered
// ---------------------------------------------------------------------------
__global__ __launch_bounds__(256)
void ggnn_main(const _Float16* __restrict__ afmT16,  // [B][NF][NN] fp16 (d_ws)
               const int*      __restrict__ bfm,     // [B][NN][NN]
               const float*    __restrict__ adj_w,   // [EF][MF][NF]
               const float*    __restrict__ bias,    // [MF]
               float*          __restrict__ out)     // [B][NN][MF]
{
  __shared__ _Float16 afmS[2][NF][JT];     // 2 x 8 KB  B-operand tiles [n][j]
  __shared__ int      bfmS[2][IT][JT];     // 2 x 2 KB  edge-type tiles
  __shared__ _Float16 aggA[IT][EF * NF];   // 32 KB     phase-2 A operand [i][k]

  const int tid  = threadIdx.x;
  const int wid  = tid >> 5;               // wave id == edge type e
  const int lane = tid & 31;
  const int lo16 = (lane < 16);
  const int l15  = lane & 15;
  const int b    = blockIdx.x / (NN / IT);
  const int I0   = (blockIdx.x % (NN / IT)) * IT;

  const _Float16* afm_b  = afmT16 + (size_t)b * NF * NN;
  const int*      bfm_bi = bfm + ((size_t)b * NN + I0) * NN;

  // warm L2/WGP$ for phase-2 weights while phase 1 runs
  __builtin_prefetch(adj_w + ((size_t)wid * MF + l15 * 8) * NF, 0, 1);

  v8f acc[NF / 16];
#pragma unroll
  for (int nb = 0; nb < NF / 16; ++nb) acc[nb] = {};
  const int e1 = wid + 1;

#if USE_TDM
  if (wid == 0) {  // TDM ignores EXEC; one issuing wave per workgroup
    tdm_load_2d((unsigned)(unsigned long long)&afmS[0][0][0], afm_b,
                NN, NF, JT, NF, NN, /*fp16*/ 1);
    tdm_load_2d((unsigned)(unsigned long long)&bfmS[0][0][0], bfm_bi,
                NN, IT, JT, IT, NN, /*i32*/ 2);
  }
#endif

  for (int c = 0; c < NCHUNK; ++c) {
    const int buf = c & 1;
    const int j0  = c * JT;

#if USE_TDM
    if (wid == 0) __builtin_amdgcn_s_wait_tensorcnt(0);  // chunk c landed
    __syncthreads();   // release consumers; previous buffer now reusable
    if (wid == 0 && c + 1 < NCHUNK) {                    // DMA chunk c+1
      const int j1 = j0 + JT;
      tdm_load_2d((unsigned)(unsigned long long)&afmS[buf ^ 1][0][0],
                  afm_b + j1, NN, NF, JT, NF, NN, 1);
      tdm_load_2d((unsigned)(unsigned long long)&bfmS[buf ^ 1][0][0],
                  bfm_bi + j1, NN, IT, JT, IT, NN, 2);
    }
#else
    __syncthreads();   // previous iteration's consumers done
    {                  // plain vectorized copies (data already fp16/contig rows)
      const int n = tid >> 1, part = (tid & 1) * 16;
      const v8h* src = reinterpret_cast<const v8h*>(afm_b + (size_t)n * NN + j0 + part);
      *reinterpret_cast<v8h*>(&afmS[buf][n][part])     = src[0];
      *reinterpret_cast<v8h*>(&afmS[buf][n][part + 8]) = src[1];
#pragma unroll
      for (int r = 0; r < 2; ++r) {
        int idx = tid + r * 256;
        bfmS[buf][idx >> 5][idx & 31] = bfm_bi[(size_t)(idx >> 5) * NN + j0 + (idx & 31)];
      }
    }
    __syncthreads();
#endif

    // ---- one-hot A tile in the 16-bit A VGPR layout --------------------
    // lane<16: row M=lane, halves 0..7 -> K 0..7, 8..15 -> K 16..23
    // lane>=16: row M=lane-16, K 8..15 and 24..31
    const int kb = lo16 ? 0 : 8;
    v16h A;
#pragma unroll
    for (int t = 0; t < 8; ++t) {
      A[t]     = (bfmS[buf][l15][kb + t]      == e1) ? (_Float16)1.0f : (_Float16)0.0f;
      A[t + 8] = (bfmS[buf][l15][kb + 16 + t] == e1) ? (_Float16)1.0f : (_Float16)0.0f;
    }

    // ---- 8 WMMAs per wave: agg tile per 16-wide n-block, A reused ------
    const int jb = lo16 ? 0 : 8;   // B layout: lanes 0-15 K=0..7,16..23; 16-31 K=8..15,24..31
#pragma unroll
    for (int nb = 0; nb < NF / 16; ++nb) {
      const int n = nb * 16 + l15;
      const v8h* bp = reinterpret_cast<const v8h*>(&afmS[buf][n][jb]);
      v16h Bt = cat16(bp[0], bp[2]);
      acc[nb] = __builtin_amdgcn_wmma_f32_16x16x32_f16(
          false, A, false, Bt, (short)0, acc[nb], false, false);
    }
  }

  // ---- spill agg (D layout -> row-major fp16 [i][e*128+n]) into LDS -----
#pragma unroll
  for (int nb = 0; nb < NF / 16; ++nb) {
    const int kcol  = wid * NF + nb * 16 + l15;
    const int rbase = lo16 ? 0 : 8;
#pragma unroll
    for (int r = 0; r < 8; ++r)
      aggA[rbase + r][kcol] = (_Float16)acc[nb][r];
  }
  __syncthreads();

  // ---- phase 2: out m-tile per wave, K = EF*NF sums over edge types -----
  v8f acc2 = {};
  const int m = wid * 16 + l15;
  for (int k0 = 0; k0 < EF * NF; k0 += 32) {
    const int aoff = lo16 ? 0 : 16;
    v8h alo = *reinterpret_cast<const v8h*>(&aggA[l15][k0 + aoff]);
    v8h ahi = *reinterpret_cast<const v8h*>(&aggA[l15][k0 + aoff + 8]);
    v16h A2 = cat16(alo, ahi);

    const int e  = k0 >> 7;
    const int n0 = (k0 & (NF - 1)) + (lo16 ? 0 : 16);
    const float4* wp = reinterpret_cast<const float4*>(
        adj_w + ((size_t)e * MF + m) * NF + n0);
    v16h B2;
#pragma unroll
    for (int q = 0; q < 4; ++q) {
      const float4 wv = wp[q];
      B2[q * 4 + 0] = (_Float16)wv.x;
      B2[q * 4 + 1] = (_Float16)wv.y;
      B2[q * 4 + 2] = (_Float16)wv.z;
      B2[q * 4 + 3] = (_Float16)wv.w;
    }
    acc2 = __builtin_amdgcn_wmma_f32_16x16x32_f16(
        false, A2, false, B2, (short)0, acc2, false, false);
  }

  const float bm = bias[m];
  const int rbase = lo16 ? 0 : 8;
#pragma unroll
  for (int r = 0; r < 8; ++r)
    out[((size_t)b * NN + I0 + rbase + r) * MF + m] = acc2[r] + bm;
}

// ---------------------------------------------------------------------------
// fallback (no workspace): round-1 kernel, converts f32->f16 while staging
// ---------------------------------------------------------------------------
__global__ __launch_bounds__(256)
void ggnn_fallback(const float* __restrict__ afm,
                   const int*   __restrict__ bfm,
                   const float* __restrict__ adj_w,
                   const float* __restrict__ bias,
                   float* __restrict__ out)
{
  __shared__ int      bfm_s[IT][JT];
  __shared__ _Float16 afmT[NF][JT];
  __shared__ _Float16 aggA[IT][EF * NF];

  const int tid  = threadIdx.x;
  const int wid  = tid >> 5;
  const int lane = tid & 31;
  const int lo16 = (lane < 16);
  const int l15  = lane & 15;
  const int b    = blockIdx.x / (NN / IT);
  const int I0   = (blockIdx.x % (NN / IT)) * IT;

  v8f acc[NF / 16];
#pragma unroll
  for (int nb = 0; nb < NF / 16; ++nb) acc[nb] = {};
  const int e1 = wid + 1;

  for (int j0 = 0; j0 < NN; j0 += JT) {
    __syncthreads();
#pragma unroll
    for (int r = 0; r < 2; ++r) {
      int idx = tid + r * 256;
      int ro = idx >> 5, co = idx & 31;
      bfm_s[ro][co] = bfm[((size_t)b * NN + I0 + ro) * NN + j0 + co];
    }
#pragma unroll
    for (int v = 0; v < 4; ++v) {
      int f4 = tid + v * 256;
      int j  = f4 >> 5;
      int c4 = f4 & 31;
      const float4 val = *reinterpret_cast<const float4*>(
          afm + ((size_t)b * NN + j0 + j) * NF + c4 * 4);
      afmT[c4 * 4 + 0][j] = (_Float16)val.x;
      afmT[c4 * 4 + 1][j] = (_Float16)val.y;
      afmT[c4 * 4 + 2][j] = (_Float16)val.z;
      afmT[c4 * 4 + 3][j] = (_Float16)val.w;
    }
    __syncthreads();

    const int kb = lo16 ? 0 : 8;
    v16h A;
#pragma unroll
    for (int t = 0; t < 8; ++t) {
      A[t]     = (bfm_s[l15][kb + t]      == e1) ? (_Float16)1.0f : (_Float16)0.0f;
      A[t + 8] = (bfm_s[l15][kb + 16 + t] == e1) ? (_Float16)1.0f : (_Float16)0.0f;
    }
    const int jb = lo16 ? 0 : 8;
#pragma unroll
    for (int nb = 0; nb < NF / 16; ++nb) {
      const int n = nb * 16 + l15;
      const v8h* bp = reinterpret_cast<const v8h*>(&afmT[n][jb]);
      v16h Bt = cat16(bp[0], bp[2]);
      acc[nb] = __builtin_amdgcn_wmma_f32_16x16x32_f16(
          false, A, false, Bt, (short)0, acc[nb], false, false);
    }
  }

#pragma unroll
  for (int nb = 0; nb < NF / 16; ++nb) {
    const int kcol  = wid * NF + nb * 16 + l15;
    const int rbase = lo16 ? 0 : 8;
#pragma unroll
    for (int r = 0; r < 8; ++r)
      aggA[rbase + r][kcol] = (_Float16)acc[nb][r];
  }
  __syncthreads();

  v8f acc2 = {};
  const int m = wid * 16 + l15;
  for (int k0 = 0; k0 < EF * NF; k0 += 32) {
    const int aoff = lo16 ? 0 : 16;
    v8h alo = *reinterpret_cast<const v8h*>(&aggA[l15][k0 + aoff]);
    v8h ahi = *reinterpret_cast<const v8h*>(&aggA[l15][k0 + aoff + 8]);
    v16h A2 = cat16(alo, ahi);
    const int e  = k0 >> 7;
    const int n0 = (k0 & (NF - 1)) + (lo16 ? 0 : 16);
    const float4* wp = reinterpret_cast<const float4*>(
        adj_w + ((size_t)e * MF + m) * NF + n0);
    v16h B2;
#pragma unroll
    for (int q = 0; q < 4; ++q) {
      const float4 wv = wp[q];
      B2[q * 4 + 0] = (_Float16)wv.x;
      B2[q * 4 + 1] = (_Float16)wv.y;
      B2[q * 4 + 2] = (_Float16)wv.z;
      B2[q * 4 + 3] = (_Float16)wv.w;
    }
    acc2 = __builtin_amdgcn_wmma_f32_16x16x32_f16(
        false, A2, false, B2, (short)0, acc2, false, false);
  }

  const float bm = bias[m];
  const int rbase = lo16 ? 0 : 8;
#pragma unroll
  for (int r = 0; r < 8; ++r)
    out[((size_t)b * NN + I0 + rbase + r) * MF + m] = acc2[r] + bm;
}

// ---------------------------------------------------------------------------
extern "C" void kernel_launch(void* const* d_in, const int* in_sizes, int n_in,
                              void* d_out, int out_size, void* d_ws, size_t ws_size,
                              hipStream_t stream) {
  const float* afm   = (const float*)d_in[0];   // [B,N,NF]
  const int*   bfm   = (const int*)  d_in[1];   // [B,N,N]
  const float* adj_w = (const float*)d_in[2];   // [EF,MF,NF]
  const float* bias  = (const float*)d_in[3];   // [MF]
  float*       out   = (float*)d_out;           // [B,N,MF]

  const size_t need = (size_t)BB * NF * NN * 2;  // fp16 transposed afm
  const dim3 grid(BB * (NN / IT)), block(256);

  if (d_ws && ws_size >= need) {
    _Float16* afmT16 = (_Float16*)d_ws;
    ggnn_prep<<<dim3((BB * NN * NF / 4) / 256), block, 0, stream>>>(afm, afmT16);
    ggnn_main<<<grid, block, 0, stream>>>(afmT16, bfm, adj_w, bias, out);
  } else {
    ggnn_fallback<<<grid, block, 0, stream>>>(afm, bfm, adj_w, bias, out);
  }
}
